// lfc_78245714199267
// MI455X (gfx1250) — 16-bit and 8-bit matrix operand order repaired and checked by reading, compiled, not run on hardware
//
#include <hip/hip_runtime.h>
#include <stdint.h>

#define BN_EPS 1e-5f

typedef __attribute__((ext_vector_type(8)))  int    v8i;
typedef __attribute__((ext_vector_type(8)))  float  v8f;
typedef __attribute__((ext_vector_type(16))) __bf16 v16bf;

__device__ __forceinline__ float  sgnf(float v){ return (v > 0.f) ? 1.f : ((v < 0.f) ? -1.f : 0.f); }
__device__ __forceinline__ int8_t sgnb(float v){ return (v > 0.f) ? (int8_t)1 : ((v < 0.f) ? (int8_t)-1 : (int8_t)0); }

__device__ __forceinline__ v8i ld_a_i8(const int8_t* p){
    int4 t0 = *(const int4*)(p);
    int4 t1 = *(const int4*)(p + 32);
    v8i a = {t0.x, t0.y, t0.z, t0.w, t1.x, t1.y, t1.z, t1.w};
    return a;
}
__device__ __forceinline__ v8i ld_b_i8(const int8_t* p){
    int4 lo = *(const int4*)(p);
    int4 hi = *(const int4*)(p + 32);
    v8i b = {lo.x, lo.y, lo.z, lo.w, hi.x, hi.y, hi.z, hi.w};
    return b;
}
__device__ __forceinline__ v16bf ld_a_bf16(const char* p){
    union { int4 i[2]; v16bf v; } u;
    u.i[0] = *(const int4*)(p);
    u.i[1] = *(const int4*)(p + 32);
    return u.v;
}
__device__ __forceinline__ v16bf ld_b_bf16(const char* p){
    union { int4 i[2]; v16bf v; } u;
    u.i[0] = *(const int4*)(p);
    u.i[1] = *(const int4*)(p + 32);
    return u.v;
}

__global__ void split_x(const float* __restrict__ X, __bf16* __restrict__ H,
                        __bf16* __restrict__ Mid, __bf16* __restrict__ L, int total){
    int i = blockIdx.x * blockDim.x + threadIdx.x;
    if (i < total){
        unsigned row = (unsigned)i / 800u, k = (unsigned)i - row * 800u;
        float v = (k < 784u) ? X[row * 784u + k] : 0.f;
        __bf16 h = (__bf16)v;  float r  = v - (float)h;
        __bf16 m = (__bf16)r;  float r2 = r - (float)m;
        __bf16 l = (__bf16)r2;
        H[i] = h; Mid[i] = m; L[i] = l;
    }
}
__global__ void bin_w1_bf16(const float* __restrict__ W, __bf16* __restrict__ Wb, int total){
    int i = blockIdx.x * blockDim.x + threadIdx.x;
    if (i < total){
        unsigned row = (unsigned)i / 800u, k = (unsigned)i - row * 800u;
        Wb[i] = (k < 784u) ? (__bf16)sgnf(W[row * 784u + k]) : (__bf16)0.f;
    }
}
__global__ void bin_w_i8(const float* __restrict__ W, int8_t* __restrict__ Wb, int n){
    int i = blockIdx.x * blockDim.x + threadIdx.x;
    if (i < n) Wb[i] = sgnb(W[i]);
}
__global__ void bin_w4_pad(const float* __restrict__ W, int8_t* __restrict__ Wb){
    int i = blockIdx.x * blockDim.x + threadIdx.x;
    if (i < 16 * 1024){
        int r = i >> 10;
        Wb[i] = (r < 10) ? sgnb(W[i]) : (int8_t)0;
    }
}

__global__ __launch_bounds__(256) void gemm1_bf16(const __bf16* __restrict__ XH,
                                                  const __bf16* __restrict__ XM,
                                                  const __bf16* __restrict__ XL,
                                                  const __bf16* __restrict__ Wb,
                                                  float* __restrict__ Y,
                                                  float* __restrict__ gSum,
                                                  float* __restrict__ gSq){
    const int KB = 1600;
    const int lane = threadIdx.x & 31;
    const int wave = threadIdx.x >> 5;
    const int half = lane >> 4;
    const int lc   = lane & 15;
    const int mBase = blockIdx.y * 128 + wave * 16;
    const int nBase = blockIdx.x * 128;

    __shared__ float ldsS[128], ldsQ[128];
    if (threadIdx.x < 128){ ldsS[threadIdx.x] = 0.f; ldsQ[threadIdx.x] = 0.f; }
    __syncthreads();

    v8f acc[8] = {};
    const unsigned aoff = (unsigned)(mBase + lc) * KB + half * 16;
    const unsigned boff = (unsigned)(nBase + lc) * KB + half * 16;
    const char* ph = (const char*)XH + aoff;
    const char* pm = (const char*)XM + aoff;
    const char* pl = (const char*)XL + aoff;
    const char* pb = (const char*)Wb + boff;

    #pragma unroll
    for (int kk = 0; kk < 25; kk++){
        const int ko = kk * 64;
        v16bf ah = ld_a_bf16(ph + ko);
        v16bf am = ld_a_bf16(pm + ko);
        v16bf al = ld_a_bf16(pl + ko);
        #pragma unroll
        for (int t = 0; t < 8; t++){
            v16bf b = ld_b_bf16(pb + t * 25600 + ko);
            acc[t] = __builtin_amdgcn_wmma_f32_16x16x32_bf16(false, al, false, b, (short)0, acc[t], false, false);
            acc[t] = __builtin_amdgcn_wmma_f32_16x16x32_bf16(false, am, false, b, (short)0, acc[t], false, false);
            acc[t] = __builtin_amdgcn_wmma_f32_16x16x32_bf16(false, ah, false, b, (short)0, acc[t], false, false);
        }
    }

    #pragma unroll
    for (int t = 0; t < 8; t++){
        float s = 0.f, q = 0.f;
        const int col = nBase + t * 16 + lc;
        #pragma unroll
        for (int r = 0; r < 8; r++){
            const int row = mBase + half * 8 + r;
            float v = acc[t][r];
            Y[(size_t)row * 1024 + col] = v;
            s += v; q += v * v;
        }
        atomicAdd(&ldsS[t * 16 + lc], s);
        atomicAdd(&ldsQ[t * 16 + lc], q);
    }
    __syncthreads();
    if (threadIdx.x < 128){
        unsafeAtomicAdd(&gSum[nBase + threadIdx.x], ldsS[threadIdx.x]);
        unsafeAtomicAdd(&gSq [nBase + threadIdx.x], ldsQ[threadIdx.x]);
    }
}

template<int NT>
__global__ __launch_bounds__(256) void gemm_i8(const int8_t* __restrict__ A,
                                               const int8_t* __restrict__ Wb,
                                               float* __restrict__ Y,
                                               float* __restrict__ gSum,
                                               float* __restrict__ gSq,
                                               int ldN){
    const int K = 1024;
    const int lane = threadIdx.x & 31;
    const int wave = threadIdx.x >> 5;
    const int half = lane >> 4;
    const int lc   = lane & 15;
    const int mBase = blockIdx.y * 256 + wave * 32;
    const int nBase = blockIdx.x * (NT * 16);

    __shared__ float ldsS[128], ldsQ[128];
    if (threadIdx.x < NT * 16){ ldsS[threadIdx.x] = 0.f; ldsQ[threadIdx.x] = 0.f; }
    __syncthreads();

    v8i acc0[NT] = {}, acc1[NT] = {};
    const int8_t* pa = A  + (unsigned)(mBase + lc) * K + half * 16;
    const int8_t* pb = Wb + (unsigned)(nBase + lc) * K + half * 16;

    #pragma unroll
    for (int kk = 0; kk < K; kk += 64){
        v8i a0 = ld_a_i8(pa + kk);
        v8i a1 = ld_a_i8(pa + 16 * K + kk);
        #pragma unroll
        for (int t = 0; t < NT; t++){
            v8i b = ld_b_i8(pb + t * 16 * K + kk);
            acc0[t] = __builtin_amdgcn_wmma_i32_16x16x64_iu8(true, a0, true, b, acc0[t], false, false);
            acc1[t] = __builtin_amdgcn_wmma_i32_16x16x64_iu8(true, a1, true, b, acc1[t], false, false);
        }
    }

    #pragma unroll
    for (int t = 0; t < NT; t++){
        float s = 0.f, q = 0.f;
        const int col = nBase + t * 16 + lc;
        #pragma unroll
        for (int r = 0; r < 8; r++){
            const int row0 = mBase + half * 8 + r;
            float v0 = (float)acc0[t][r];
            Y[(size_t)row0 * ldN + col] = v0; s += v0; q += v0 * v0;
            const int row1 = row0 + 16;
            float v1 = (float)acc1[t][r];
            Y[(size_t)row1 * ldN + col] = v1; s += v1; q += v1 * v1;
        }
        atomicAdd(&ldsS[t * 16 + lc], s);
        atomicAdd(&ldsQ[t * 16 + lc], q);
    }
    __syncthreads();
    if (threadIdx.x < NT * 16){
        unsafeAtomicAdd(&gSum[nBase + threadIdx.x], ldsS[threadIdx.x]);
        unsafeAtomicAdd(&gSq [nBase + threadIdx.x], ldsQ[threadIdx.x]);
    }
}

__global__ void bn_finalize(const float* __restrict__ gSum, const float* __restrict__ gSq,
                            const float* __restrict__ g, const float* __restrict__ be,
                            float* __restrict__ aArr, float* __restrict__ cArr,
                            int n, float invB){
    int i = blockIdx.x * blockDim.x + threadIdx.x;
    if (i < n){
        float mu  = gSum[i] * invB;
        float var = gSq[i] * invB - mu * mu;
        float rs  = rsqrtf(var + BN_EPS);
        float a   = g[i] * rs;
        aArr[i] = a;
        cArr[i] = be[i] - mu * a;
    }
}

__global__ void bin_act(const float* __restrict__ Y, const float* __restrict__ aArr,
                        const float* __restrict__ cArr, int8_t* __restrict__ S, int total){
    int i = blockIdx.x * blockDim.x + threadIdx.x;
    int idx = i * 4;
    if (idx < total){
        float4 y = *(const float4*)(Y + idx);
        int c0 = idx & 1023;
        char4 o;
        o.x = sgnb(y.x * aArr[c0]     + cArr[c0]);
        o.y = sgnb(y.y * aArr[c0 + 1] + cArr[c0 + 1]);
        o.z = sgnb(y.z * aArr[c0 + 2] + cArr[c0 + 2]);
        o.w = sgnb(y.w * aArr[c0 + 3] + cArr[c0 + 3]);
        *(char4*)(S + idx) = o;
    }
}

__global__ void final_out(const float* __restrict__ Y4, const float* __restrict__ aArr,
                          const float* __restrict__ cArr, float* __restrict__ out, int M){
    int i = blockIdx.x * blockDim.x + threadIdx.x;
    if (i < M * 10){
        int m = i / 10, n = i - m * 10;
        out[i] = Y4[(size_t)m * 16 + n] * aArr[n] + cArr[n];
    }
}

extern "C" void kernel_launch(void* const* d_in, const int* in_sizes, int n_in,
                              void* d_out, int out_size, void* d_ws, size_t ws_size,
                              hipStream_t stream){
    const int B = 32768;
    const float invB = 1.f / (float)B;

    const float* x   = (const float*)d_in[0];
    const float* W1  = (const float*)d_in[1];
    const float* g1  = (const float*)d_in[3];
    const float* be1 = (const float*)d_in[4];
    const float* W2  = (const float*)d_in[5];
    const float* g2  = (const float*)d_in[7];
    const float* be2 = (const float*)d_in[8];
    const float* W3  = (const float*)d_in[9];
    const float* g3  = (const float*)d_in[11];
    const float* be3 = (const float*)d_in[12];
    const float* W4  = (const float*)d_in[13];
    const float* g4  = (const float*)d_in[15];
    const float* be4 = (const float*)d_in[16];
    float* out = (float*)d_out;

    char* ws = (char*)d_ws;
    float*  yBuf = (float*)ws;   ws += (size_t)B * 1024 * 4;
    __bf16* xh   = (__bf16*)ws;  ws += (size_t)B * 800 * 2;
    int8_t* sBuf = (int8_t*)xh;
    __bf16* xm   = (__bf16*)ws;  ws += (size_t)B * 800 * 2;
    __bf16* xl   = (__bf16*)ws;  ws += (size_t)B * 800 * 2;
    __bf16* w1b  = (__bf16*)ws;  ws += (size_t)1024 * 800 * 2;
    int8_t* w2b  = (int8_t*)ws;  ws += (size_t)1024 * 1024;
    int8_t* w3b  = (int8_t*)ws;  ws += (size_t)1024 * 1024;
    int8_t* w4b  = (int8_t*)ws;  ws += (size_t)16 * 1024;
    float*  gSum = (float*)ws;   ws += 4096;
    float*  gSq  = (float*)ws;   ws += 4096;
    float*  aArr = (float*)ws;   ws += 4096;
    float*  cArr = (float*)ws;   ws += 4096;

    split_x    <<<(B * 800 + 255) / 256, 256, 0, stream>>>(x, xh, xm, xl, B * 800);
    bin_w1_bf16<<<(1024 * 800 + 255) / 256, 256, 0, stream>>>(W1, w1b, 1024 * 800);
    bin_w_i8   <<<(1024 * 1024 + 255) / 256, 256, 0, stream>>>(W2, w2b, 1024 * 1024);
    bin_w_i8   <<<(1024 * 1024 + 255) / 256, 256, 0, stream>>>(W3, w3b, 1024 * 1024);
    bin_w4_pad <<<(16 * 1024 + 255) / 256, 256, 0, stream>>>(W4, w4b);

    dim3 blk(256);
    const int actTotal = B * 1024;
    const int actGrid  = (actTotal / 4 + 255) / 256;

    hipMemsetAsync(gSum, 0, 8192, stream);
    gemm1_bf16<<<dim3(8, 256), blk, 0, stream>>>(xh, xm, xl, w1b, yBuf, gSum, gSq);
    bn_finalize<<<4, 256, 0, stream>>>(gSum, gSq, g1, be1, aArr, cArr, 1024, invB);
    bin_act<<<actGrid, 256, 0, stream>>>(yBuf, aArr, cArr, sBuf, actTotal);

    hipMemsetAsync(gSum, 0, 8192, stream);
    gemm_i8<8><<<dim3(8, 128), blk, 0, stream>>>(sBuf, w2b, yBuf, gSum, gSq, 1024);
    bn_finalize<<<4, 256, 0, stream>>>(gSum, gSq, g2, be2, aArr, cArr, 1024, invB);
    bin_act<<<actGrid, 256, 0, stream>>>(yBuf, aArr, cArr, sBuf, actTotal);

    hipMemsetAsync(gSum, 0, 8192, stream);
    gemm_i8<8><<<dim3(8, 128), blk, 0, stream>>>(sBuf, w3b, yBuf, gSum, gSq, 1024);
    bn_finalize<<<4, 256, 0, stream>>>(gSum, gSq, g3, be3, aArr, cArr, 1024, invB);
    bin_act<<<actGrid, 256, 0, stream>>>(yBuf, aArr, cArr, sBuf, actTotal);

    hipMemsetAsync(gSum, 0, 8192, stream);
    gemm_i8<1><<<dim3(1, 128), blk, 0, stream>>>(sBuf, w4b, yBuf, gSum, gSq, 16);
    bn_finalize<<<1, 256, 0, stream>>>(gSum, gSq, g4, be4, aArr, cArr, 16, invB);
    final_out<<<(B * 10 + 255) / 256, 256, 0, stream>>>(yBuf, aArr, cArr, out, B);
}
